// MultiScaleRetention_85126251807267
// MI455X (gfx1250) — matrix operand order repaired and checked by reading, not compiled, not run on hardware
//
#include <hip/hip_runtime.h>
#include <hip/hip_bf16.h>

// ---------------------------------------------------------------------------
// Multi-scale retention (RetNet) forward for MI455X (gfx1250, wave32, WMMA).
// B=2, L=4096, D=1024, H=8, head_dim=128, chunk=64.
// ---------------------------------------------------------------------------

#define BB   2
#define LL   4096
#define DD   1024
#define HH   8
#define HD   128
#define CH   64
#define NC   (LL / CH)          // 64 chunks
#define MM   (BB * LL)          // 8192 rows

typedef __attribute__((ext_vector_type(16))) __bf16 v16bf;
typedef __attribute__((ext_vector_type(8)))  __bf16 v8bf;
typedef __attribute__((ext_vector_type(8)))  float  v8f;

// ---- async global->LDS path (gfx1250), guarded so either toolchain builds --
#if defined(__has_builtin)
#if __has_builtin(__builtin_amdgcn_global_load_async_to_lds_b128) && \
    __has_builtin(__builtin_amdgcn_s_wait_asynccnt)
#define HAVE_ASYNC_LDS 1
#endif
#endif

#if defined(HAVE_ASYNC_LDS)
typedef int v4i_vs __attribute__((vector_size(16)));
static __device__ __forceinline__ void async_cp128(void* lds, const void* g) {
  __builtin_amdgcn_global_load_async_to_lds_b128(
      (__attribute__((address_space(1))) v4i_vs*)g,
      (__attribute__((address_space(3))) v4i_vs*)lds,
      0, 0);
}
static __device__ __forceinline__ void async_wait0() {
  __builtin_amdgcn_s_wait_asynccnt(0);
}
#endif

static __device__ __forceinline__ v8f wmma_bf16(v16bf a, v16bf b, v8f c) {
  return __builtin_amdgcn_wmma_f32_16x16x32_bf16(false, a, false, b, (short)0, c,
                                                 false, false);
}

static __device__ __forceinline__ v16bf cat8(v8bf lo, v8bf hi) {
  return __builtin_shufflevector(lo, hi, 0, 1, 2, 3, 4, 5, 6, 7,
                                         8, 9, 10, 11, 12, 13, 14, 15);
}

// A fragment 16x32 from row-major bf16 A[row][k], row stride ld (mult of 8).
static __device__ __forceinline__ v16bf frag_ld_a(const __bf16* p, int ld) {
  const int lane = threadIdx.x & 31;
  const int row  = lane & 15;
  const int kh   = (lane >> 4) & 1;
  const __bf16* r = p + row * ld + kh * 8;
  v8bf lo = *(const v8bf*)(r);
  v8bf hi = *(const v8bf*)(r + 16);
  return cat8(lo, hi);
}

// B fragment 32x16 from K-contiguous storage Bt[n][k], row stride ld (mult 8).
static __device__ __forceinline__ v16bf frag_ld_bT(const __bf16* p, int ld) {
  const int lane = threadIdx.x & 31;
  const int col  = lane & 15;
  const int kg   = (lane >> 4) & 1;
  const __bf16* r = p + col * ld + kg * 8;
  v8bf lo = *(const v8bf*)(r);
  v8bf hi = *(const v8bf*)(r + 16);
  return cat8(lo, hi);
}

// ---------------------------------------------------------------------------
// Kernel 0a: fp32 -> bf16 (activations)
// ---------------------------------------------------------------------------
__global__ void cvt_f32_bf16(const float* __restrict__ in, __bf16* __restrict__ out,
                             int n) {
  int i = (blockIdx.x * blockDim.x + threadIdx.x) * 4;
  if (i < n) {
    float4 f = *(const float4*)(in + i);
    out[i + 0] = (__bf16)f.x;
    out[i + 1] = (__bf16)f.y;
    out[i + 2] = (__bf16)f.z;
    out[i + 3] = (__bf16)f.w;
  }
}

// ---------------------------------------------------------------------------
// Kernel 0b: fp32 -> bf16 transposed (weights): out[n][k] = in[k][n], DD x DD
// ---------------------------------------------------------------------------
__global__ __launch_bounds__(256)
void cvt_transpose_bf16(const float* __restrict__ in, __bf16* __restrict__ out) {
  __shared__ float tile[32][33];
  const int bx = blockIdx.x * 32;              // n block
  const int by = blockIdx.y * 32;              // k block
  const int tx = threadIdx.x & 31;
  const int ty = threadIdx.x >> 5;             // 0..7
#pragma unroll
  for (int i = 0; i < 32; i += 8)
    tile[ty + i][tx] = in[(size_t)(by + ty + i) * DD + bx + tx];
  __syncthreads();
#pragma unroll
  for (int i = 0; i < 32; i += 8)
    out[(size_t)(bx + ty + i) * DD + by + tx] = (__bf16)tile[tx][ty + i];
}

// ---------------------------------------------------------------------------
// Kernel 1/5: bf16 GEMM  C(MxN) = A(MxK) @ W(KxN) with W given TRANSPOSED
// (Wt[n][k]).  128x128 block tile, 8 waves * (2x4) WMMA tiles, K staged 32 at
// a time, double-buffered LDS; staging uses the async LDS-DMA path when the
// toolchain exposes it (ASYNCcnt), else a register-staged fallback.
// ---------------------------------------------------------------------------
#define ALD 40   // padded row stride for A tile (128 x 32)
#define BLD 40   // padded row stride for Bt tile (128 n x 32 k)

template <typename OutT>
__global__ __launch_bounds__(256)
void gemm_bf16_128x128(const __bf16* __restrict__ A, const __bf16* __restrict__ Wt,
                       OutT* __restrict__ C, int M, int N, int K) {
  __shared__ alignas(16) __bf16 As[2][128 * ALD];
  __shared__ alignas(16) __bf16 Bs[2][128 * BLD];

  const int m0 = blockIdx.y * 128;
  const int n0 = blockIdx.x * 128;
  const int t  = threadIdx.x;
  const int w  = t >> 5;
  const int wm = w >> 1;   // 0..3 -> 32 rows
  const int wn = w & 1;    // 0..1 -> 64 cols

  const int arow = t >> 1, acol = (t & 1) * 16;   // A: 128x32, 16 elems/thread
  const int brow = t >> 1, bcol = (t & 1) * 16;   // Bt: 128x32, 16 elems/thread

  const __bf16* aptr = A + (size_t)(m0 + arow) * K + acol;
  const __bf16* bptr = Wt + (size_t)(n0 + brow) * K + bcol;
  __bf16* asl = &As[0][arow * ALD + acol];
  __bf16* bsl = &Bs[0][brow * BLD + bcol];
  const int bufstride = 128 * ALD;

  v8f acc[2][4];
#pragma unroll
  for (int i = 0; i < 2; ++i)
#pragma unroll
    for (int j = 0; j < 4; ++j)
#pragma unroll
      for (int r = 0; r < 8; ++r) acc[i][j][r] = 0.0f;

  // prologue: stage tile 0
#if defined(HAVE_ASYNC_LDS)
  {
    async_cp128(asl, aptr);     async_cp128(asl + 8, aptr + 8);
    async_cp128(bsl, bptr);     async_cp128(bsl + 8, bptr + 8);
    async_wait0();
  }
#else
  {
    uint4 a0 = ((const uint4*)aptr)[0];
    uint4 a1 = ((const uint4*)aptr)[1];
    uint4 b0 = ((const uint4*)bptr)[0];
    uint4 b1 = ((const uint4*)bptr)[1];
    ((uint4*)asl)[0] = a0; *(uint4*)(asl + 8) = a1;
    ((uint4*)bsl)[0] = b0; *(uint4*)(bsl + 8) = b1;
  }
#endif
  __syncthreads();

  int buf = 0;
  for (int k0 = 0; k0 < K; k0 += 32) {
    const bool has_next = (k0 + 32) < K;
#if defined(HAVE_ASYNC_LDS)
    if (has_next) {                       // LDS-DMA for tile k0+32, other buffer
      const __bf16* ap = aptr + k0 + 32;
      const __bf16* bp = bptr + k0 + 32;
      __bf16* ad = asl + (buf ^ 1) * bufstride;
      __bf16* bd = bsl + (buf ^ 1) * bufstride;
      async_cp128(ad, ap); async_cp128(ad + 8, ap + 8);
      async_cp128(bd, bp); async_cp128(bd + 8, bp + 8);
    }
#else
    uint4 a0, a1, b0, b1;
    if (has_next) {                       // global loads for tile k0+32
      const __bf16* ap = aptr + k0 + 32;
      const __bf16* bp = bptr + k0 + 32;
      a0 = ((const uint4*)ap)[0]; a1 = ((const uint4*)ap)[1];
      b0 = ((const uint4*)bp)[0]; b1 = ((const uint4*)bp)[1];
    }
#endif

    // compute on current buffer (overlaps the outstanding copies)
    const __bf16* ab = As[buf];
    const __bf16* bb = Bs[buf];
    v16bf af0 = frag_ld_a(&ab[(wm * 32 + 0) * ALD], ALD);
    v16bf af1 = frag_ld_a(&ab[(wm * 32 + 16) * ALD], ALD);
#pragma unroll
    for (int j = 0; j < 4; ++j) {
      v16bf bf = frag_ld_bT(&bb[(wn * 64 + j * 16) * BLD], BLD);
      acc[0][j] = wmma_bf16(af0, bf, acc[0][j]);
      acc[1][j] = wmma_bf16(af1, bf, acc[1][j]);
    }

    if (has_next) {
#if defined(HAVE_ASYNC_LDS)
      async_wait0();
#else
      __bf16* ad = asl + (buf ^ 1) * bufstride;
      __bf16* bd = bsl + (buf ^ 1) * bufstride;
      ((uint4*)ad)[0] = a0; *(uint4*)(ad + 8) = a1;
      ((uint4*)bd)[0] = b0; *(uint4*)(bd + 8) = b1;
#endif
    }
    __syncthreads();
    buf ^= 1;
  }

  const int lane  = t & 31;
  const int rbase = (lane >> 4) * 8;
  const int cc    = lane & 15;
#pragma unroll
  for (int i = 0; i < 2; ++i)
#pragma unroll
    for (int j = 0; j < 4; ++j)
#pragma unroll
      for (int r = 0; r < 8; ++r) {
        int row = m0 + wm * 32 + i * 16 + rbase + r;
        int col = n0 + wn * 64 + j * 16 + cc;
        C[(size_t)row * N + col] = (OutT)acc[i][j][r];
      }
}

// ---------------------------------------------------------------------------
// Kernel 2: per-chunk KVt[e][d] = sum_j v[j][e] * (k[j][d]*decay_k[j])
// (i.e. KV^T = v^T @ k_scaled).  grid = (NC, B*H), 256 threads (8 waves);
// wave w owns the 16-wide d-strip.
// ---------------------------------------------------------------------------
#define TLD 72   // padded stride for 128 x 64 transposed chunk tiles

__global__ __launch_bounds__(256)
void chunk_kv_kernel(const __bf16* __restrict__ k, const __bf16* __restrict__ v,
                     __bf16* __restrict__ kv) {
  __shared__ alignas(16) __bf16 kT[HD * TLD];   // k_scaled^T : [d][j]
  __shared__ alignas(16) __bf16 vT[HD * TLD];   // v^T        : [e][j]

  const int n  = blockIdx.x;
  const int bh = blockIdx.y;
  const int b  = bh >> 3, h = bh & 7;
  const float gamma = 1.0f - exp2f(-5.0f - (float)h);

  const int t    = threadIdx.x;
  const int row  = t >> 2;            // chunk position j: 0..63
  const int coff = (t & 3) * 32;      // feature offset
  const int l    = n * CH + row;
  const size_t g = ((size_t)(b * LL + l)) * DD + h * HD + coff;
  const float dk = powf(gamma, (float)(CH - 1 - row));
#pragma unroll 8
  for (int i = 0; i < 32; ++i) {
    kT[(coff + i) * TLD + row] = (__bf16)((float)k[g + i] * dk);
    vT[(coff + i) * TLD + row] = v[g + i];
  }
  __syncthreads();

  const int w = t >> 5;
  v8f acc[8];
#pragma unroll
  for (int m = 0; m < 8; ++m)
#pragma unroll
    for (int r = 0; r < 8; ++r) acc[m][r] = 0.0f;

#pragma unroll
  for (int ks = 0; ks < 2; ++ks) {                     // K = 64 chunk positions
    v16bf bf = frag_ld_bT(&kT[(w * 16) * TLD + ks * 32], TLD);
#pragma unroll
    for (int mt = 0; mt < 8; ++mt) {                   // 128 e-rows
      v16bf af = frag_ld_a(&vT[(mt * 16) * TLD + ks * 32], TLD);
      acc[mt] = wmma_bf16(af, bf, acc[mt]);
    }
  }

  __bf16* out = kv + ((size_t)bh * NC + n) * (HD * HD);
  const int lane = t & 31;
  const int rbase = (lane >> 4) * 8;
  const int cc = lane & 15;
#pragma unroll
  for (int mt = 0; mt < 8; ++mt)
#pragma unroll
    for (int r = 0; r < 8; ++r)
      out[(mt * 16 + rbase + r) * HD + w * 16 + cc] = (__bf16)acc[mt][r];
}

// ---------------------------------------------------------------------------
// Kernel 3: decay scan over chunk states per (b,h):
//   S_0 = 0 ; store S_n ; S_{n+1} = gamma^CH * S_n + KV_n   (elementwise)
// ---------------------------------------------------------------------------
__global__ __launch_bounds__(256)
void state_scan_kernel(const __bf16* __restrict__ kv, __bf16* __restrict__ s) {
  const int bh = blockIdx.x;
  const int h  = bh & 7;
  const float gamma = 1.0f - exp2f(-5.0f - (float)h);
  const float gC = powf(gamma, (float)CH);
  const int t = threadIdx.x;

  float S[64];
#pragma unroll
  for (int e = 0; e < 64; ++e) S[e] = 0.0f;

  const __bf16* kvb = kv + (size_t)bh * NC * (HD * HD);
  __bf16* sb = s + (size_t)bh * NC * (HD * HD);
  for (int n = 0; n < NC; ++n) {
    const size_t base = (size_t)n * (HD * HD);
#pragma unroll
    for (int e = 0; e < 64; ++e) {
      const size_t idx = base + (size_t)e * 256 + t;
      sb[idx] = (__bf16)S[e];
      S[e] = gC * S[e] + (float)kvb[idx];
    }
  }
}

// ---------------------------------------------------------------------------
// Kernel 4: per-chunk output:
//   attn = mask(q_s @ k^T) ; o = attn @ v + (q_s * decay_q) @ S_n
// grid = (NC, B*H), 256 threads (8 waves).
// S is stored transposed (St[e][d]) so it is directly B-fragment friendly.
// ---------------------------------------------------------------------------
#define QLD 136  // padded stride for 64 x 128 row-major chunk tiles
#define ATLD 72  // padded stride for 64 x 64 attn tile

__global__ __launch_bounds__(256)
void chunk_out_kernel(const __bf16* __restrict__ q, const __bf16* __restrict__ k,
                      const __bf16* __restrict__ v, const __bf16* __restrict__ s,
                      __bf16* __restrict__ o) {
  __shared__ alignas(16) __bf16 qA[CH * QLD];    // q * DK^-0.5        [j][d]
  __shared__ alignas(16) __bf16 qB[CH * QLD];    // qA * gamma^(j+1)   [j][d]
  __shared__ alignas(16) __bf16 kds[CH * QLD];   // k chunk            [j][d]
  __shared__ alignas(16) __bf16 vT[HD * TLD];    // v^T                [e][j]
  __shared__ alignas(16) __bf16 Sds[HD * QLD];   // St                 [e][d]
  __shared__ alignas(16) __bf16 attn[CH * ATLD]; // masked attn        [i][j]
  __shared__ float gpow[CH];                     // gamma^0..gamma^63

  const int n  = blockIdx.x;
  const int bh = blockIdx.y;
  const int b  = bh >> 3, h = bh & 7;
  const float gamma = 1.0f - exp2f(-5.0f - (float)h);
  const float scale = 0.08838834764831845f;      // 128^-0.5

  const int t    = threadIdx.x;
  if (t < CH) gpow[t] = powf(gamma, (float)t);

  // S tile: contiguous rows -> async LDS-DMA when available
  {
    const __bf16* sbase = s + ((size_t)bh * NC + n) * (HD * HD);
    const int e = t >> 1, doff = (t & 1) * 64;
#if defined(HAVE_ASYNC_LDS)
#pragma unroll
    for (int i = 0; i < 8; ++i)
      async_cp128(&Sds[e * QLD + doff + i * 8], sbase + e * HD + doff + i * 8);
#else
    const uint4* src = (const uint4*)(sbase + e * HD + doff);
    uint4* dst = (uint4*)(&Sds[e * QLD + doff]);
#pragma unroll
    for (int i = 0; i < 8; ++i) dst[i] = src[i];
#endif
  }

  const int row  = t >> 2;
  const int coff = (t & 3) * 32;
  const int l    = n * CH + row;
  const size_t g = ((size_t)(b * LL + l)) * DD + h * HD + coff;
  const float dq = powf(gamma, (float)(row + 1));
#pragma unroll 8
  for (int i = 0; i < 32; ++i) {
    float qv = (float)q[g + i] * scale;
    qA[row * QLD + coff + i] = (__bf16)qv;
    qB[row * QLD + coff + i] = (__bf16)(qv * dq);
    kds[row * QLD + coff + i] = k[g + i];
    vT[(coff + i) * TLD + row] = v[g + i];      // transposed fill
  }
#if defined(HAVE_ASYNC_LDS)
  async_wait0();
#endif
  __syncthreads();

  const int w     = t >> 5;
  const int mt    = w >> 1;           // 0..3 : 16-row strip of the chunk
  const int nhalf = w & 1;            // 0..1 : 64-col half of head dim
  const int lane  = t & 31;
  const int rbase = (lane >> 4) * 8;
  const int cc    = lane & 15;

  // ---- phase A: attn strip (even waves; odd waves idle) ----
  if (nhalf == 0) {
    v8f a[4];
#pragma unroll
    for (int nt = 0; nt < 4; ++nt)
#pragma unroll
      for (int r = 0; r < 8; ++r) a[nt][r] = 0.0f;
#pragma unroll
    for (int ks = 0; ks < 4; ++ks) {
      v16bf af = frag_ld_a(&qA[(mt * 16) * QLD + ks * 32], QLD);
#pragma unroll
      for (int nt = 0; nt < 4; ++nt) {
        v16bf bf = frag_ld_bT(&kds[(nt * 16) * QLD + ks * 32], QLD);
        a[nt] = wmma_bf16(af, bf, a[nt]);
      }
    }
#pragma unroll
    for (int nt = 0; nt < 4; ++nt)
#pragma unroll
      for (int r = 0; r < 8; ++r) {
        int i = mt * 16 + rbase + r;       // query position
        int j = nt * 16 + cc;              // key position
        float m = (i >= j) ? gpow[i - j] : 0.0f;
        attn[i * ATLD + j] = (__bf16)(a[nt][r] * m);
      }
  }
  __syncthreads();

  // ---- phase B: o = attn @ v + qB @ S ----
  v8f acc[4];
#pragma unroll
  for (int j = 0; j < 4; ++j)
#pragma unroll
    for (int r = 0; r < 8; ++r) acc[j][r] = 0.0f;

#pragma unroll
  for (int ks = 0; ks < 2; ++ks) {         // intra: K = 64
    v16bf af = frag_ld_a(&attn[(mt * 16) * ATLD + ks * 32], ATLD);
#pragma unroll
    for (int j = 0; j < 4; ++j) {
      v16bf bf = frag_ld_bT(&vT[((nhalf * 4 + j) * 16) * TLD + ks * 32], TLD);
      acc[j] = wmma_bf16(af, bf, acc[j]);
    }
  }
#pragma unroll
  for (int ks = 0; ks < 4; ++ks) {         // inter: K = 128
    v16bf af = frag_ld_a(&qB[(mt * 16) * QLD + ks * 32], QLD);
#pragma unroll
    for (int j = 0; j < 4; ++j) {
      v16bf bf = frag_ld_bT(&Sds[((nhalf * 4 + j) * 16) * QLD + ks * 32], QLD);
      acc[j] = wmma_bf16(af, bf, acc[j]);
    }
  }

#pragma unroll
  for (int j = 0; j < 4; ++j)
#pragma unroll
    for (int r = 0; r < 8; ++r) {
      int lrow = n * CH + mt * 16 + rbase + r;
      int col  = h * HD + nhalf * 64 + j * 16 + cc;
      o[((size_t)(b * LL + lrow)) * DD + col] = (__bf16)acc[j][r];
    }
}

// ---------------------------------------------------------------------------
// Launch
// ---------------------------------------------------------------------------
extern "C" void kernel_launch(void* const* d_in, const int* in_sizes, int n_in,
                              void* d_out, int out_size, void* d_ws, size_t ws_size,
                              hipStream_t stream) {
  (void)in_sizes; (void)n_in; (void)out_size; (void)ws_size;
  const float* X  = (const float*)d_in[0];
  const float* Wq = (const float*)d_in[2];
  const float* Wk = (const float*)d_in[3];
  const float* Wv = (const float*)d_in[4];
  const float* Wo = (const float*)d_in[5];
  float* out = (float*)d_out;

  char* ws = (char*)d_ws;
  const size_t SZ_X = (size_t)MM * DD * 2;                 // bf16 activations
  const size_t SZ_W = (size_t)DD * DD * 2;                 // bf16 weights
  const size_t SZ_S = (size_t)BB * HH * NC * HD * HD * 2;  // bf16 states

  __bf16* Xb  = (__bf16*)(ws);
  __bf16* Wqt = (__bf16*)(ws + SZ_X);
  __bf16* Wkt = (__bf16*)(ws + SZ_X + SZ_W);
  __bf16* Wvt = (__bf16*)(ws + SZ_X + 2 * SZ_W);
  __bf16* Wot = (__bf16*)(ws + SZ_X + 3 * SZ_W);
  __bf16* Qb  = (__bf16*)(ws + SZ_X + 4 * SZ_W);
  __bf16* Kb  = (__bf16*)(ws + 2 * SZ_X + 4 * SZ_W);
  __bf16* Vb  = (__bf16*)(ws + 3 * SZ_X + 4 * SZ_W);
  __bf16* Ob  = (__bf16*)(ws + 4 * SZ_X + 4 * SZ_W);
  __bf16* KVb = (__bf16*)(ws + 5 * SZ_X + 4 * SZ_W);
  __bf16* Sb  = (__bf16*)(ws + 5 * SZ_X + 4 * SZ_W + SZ_S);

  // 0) conversions: X plain, weights transposed (Wt[n][k])
  {
    int nX = MM * DD;
    cvt_f32_bf16<<<nX / (256 * 4), 256, 0, stream>>>(X, Xb, nX);
    dim3 tg(DD / 32, DD / 32);
    cvt_transpose_bf16<<<tg, 256, 0, stream>>>(Wq, Wqt);
    cvt_transpose_bf16<<<tg, 256, 0, stream>>>(Wk, Wkt);
    cvt_transpose_bf16<<<tg, 256, 0, stream>>>(Wv, Wvt);
    cvt_transpose_bf16<<<tg, 256, 0, stream>>>(Wo, Wot);
  }

  // 1) Q/K/V projections: (8192x1024) @ (1024x1024), bf16 out
  dim3 gg(DD / 128, MM / 128);
  gemm_bf16_128x128<__bf16><<<gg, 256, 0, stream>>>(Xb, Wqt, Qb, MM, DD, DD);
  gemm_bf16_128x128<__bf16><<<gg, 256, 0, stream>>>(Xb, Wkt, Kb, MM, DD, DD);
  gemm_bf16_128x128<__bf16><<<gg, 256, 0, stream>>>(Xb, Wvt, Vb, MM, DD, DD);

  // 2) per-chunk KV^T outer products
  chunk_kv_kernel<<<dim3(NC, BB * HH), 256, 0, stream>>>(Kb, Vb, KVb);

  // 3) decay scan over chunk states
  state_scan_kernel<<<BB * HH, 256, 0, stream>>>(KVb, Sb);

  // 4) per-chunk retention output
  chunk_out_kernel<<<dim3(NC, BB * HH), 256, 0, stream>>>(Qb, Kb, Vb, Sb, Ob);

  // 5) output projection: (8192x1024) @ (1024x1024) -> fp32
  gemm_bf16_128x128<float><<<gg, 256, 0, stream>>>(Ob, Wot, out, MM, DD, DD);
}
